// VisionMamba3D_60258391163325
// MI455X (gfx1250) — matrix operand order repaired and checked by reading, not compiled, not run on hardware
//
#include <hip/hip_runtime.h>
#include <hip/hip_bf16.h>
#include <math.h>

#define D_MODEL 192
#define D_INNER 384
#define N_STATE 16
#define D_CONV  4
#define DT_RANK 12
#define DEPTH   24
#define LTOK    512       // 8x8x8 patches
#define PATCH_ELEMS 4096  // 16^3
#define XPAD    48        // 44 padded to 48 for 16-wide WMMA tiles

typedef __attribute__((ext_vector_type(16))) _Float16 v16h;
typedef __attribute__((ext_vector_type(8)))  _Float16 v8h;
typedef __attribute__((ext_vector_type(8)))  float    v8f;

// ---------------------------------------------------------------------------
// Generic helpers
// ---------------------------------------------------------------------------
__global__ void k_cvt_f32_f16(const float* __restrict__ s, _Float16* __restrict__ d, int n) {
    int i = blockIdx.x * blockDim.x + threadIdx.x;
    if (i < n) d[i] = (_Float16)s[i];
}

__global__ void k_zero_f32(float* __restrict__ d, int n) {
    int i = blockIdx.x * blockDim.x + threadIdx.x;
    if (i < n) d[i] = 0.f;
}

// Pad x_proj weights (DEPTH,44,384) -> (DEPTH,48,384) f16, zero rows 44..47
__global__ void k_pad_xproj(const float* __restrict__ src, _Float16* __restrict__ dst) {
    int i = blockIdx.x * blockDim.x + threadIdx.x;
    int total = DEPTH * XPAD * D_INNER;
    if (i >= total) return;
    int k = i % D_INNER;
    int r = (i / D_INNER) % XPAD;
    int l = i / (XPAD * D_INNER);
    float v = (r < DT_RANK + 2 * N_STATE) ? src[(l * 44 + r) * D_INNER + k] : 0.f;
    dst[i] = (_Float16)v;
}

// im2col: x (1,1,128,128,128) -> A0 (512 x 4096) f16
__global__ void k_im2col(const float* __restrict__ x, _Float16* __restrict__ A0) {
    int t  = blockIdx.x;                 // token 0..511
    int pd = t >> 6, ph = (t >> 3) & 7, pw = t & 7;
    for (int e = threadIdx.x; e < PATCH_ELEMS; e += blockDim.x) {
        int kd = e >> 8, kh = (e >> 4) & 15, kw = e & 15;
        int xi = ((pd * 16 + kd) * 128 + (ph * 16 + kh)) * 128 + (pw * 16 + kw);
        A0[(size_t)t * PATCH_ELEMS + e] = (_Float16)x[xi];
    }
}

// ---------------------------------------------------------------------------
// WMMA GEMM: C[M, N](f32) = A[M,K](f16 row-major) * Bt[N,K](f16 row-major)^T
// One wave32 per 16 x (16*NT) output strip. Per k-step: A fragment + all NT
// B fragments loaded into distinct live regs FIRST (one big load clause,
// partial s_wait_loadcnt), then NT back-to-back WMMAs reusing A.
// Fragment layouts per CDNA5 ISA 7.12.2. K % 32 == 0, M % 16 == 0,
// N % (16*NT) == 0.
// ---------------------------------------------------------------------------
template <int NT>
__global__ void k_wmma_gemm(const _Float16* __restrict__ A,
                            const _Float16* __restrict__ Bt,
                            float* __restrict__ C,
                            const float* __restrict__ bias,
                            int M, int N, int K) {
    int lane = threadIdx.x;            // 0..31, EXEC all ones
    int tn0  = blockIdx.x * NT;        // first N tile of this strip
    int tm   = blockIdx.y;             // M tile
    int mn   = lane & 15;
    int hiA  = (lane >> 4) << 3;       // K phase for A: 0 or 8
    int hiB  = (lane >> 4) << 3;       // K phase for B: 0 or 8
    const _Float16* arow = A + (size_t)(tm * 16 + mn) * K;
    const _Float16* brow[NT];
#pragma unroll
    for (int j = 0; j < NT; j++)
        brow[j] = Bt + (size_t)((tn0 + j) * 16 + mn) * K;

    v8f acc[NT];
#pragma unroll
    for (int j = 0; j < NT; j++) acc[j] = (v8f){0.f,0.f,0.f,0.f,0.f,0.f,0.f,0.f};

    for (int k0 = 0; k0 < K; k0 += 32) {
        // ---- load phase: issue everything, keep all fragments live ----
        v8h a0 = *(const v8h*)(arow + k0 + hiA);
        v8h a1 = *(const v8h*)(arow + k0 + hiA + 16);
        v8h b0[NT], b1[NT];
#pragma unroll
        for (int j = 0; j < NT; j++) {
            b0[j] = *(const v8h*)(brow[j] + k0 + hiB);
            b1[j] = *(const v8h*)(brow[j] + k0 + hiB + 16);
        }
        v16h a;
#pragma unroll
        for (int i = 0; i < 8; i++) { a[i] = a0[i]; a[i + 8] = a1[i]; }
        // ---- compute phase: NT WMMAs reuse the A fragment ----
#pragma unroll
        for (int j = 0; j < NT; j++) {
            v16h b;
#pragma unroll
            for (int i = 0; i < 8; i++) { b[i] = b0[j][i]; b[i + 8] = b1[j][i]; }
            acc[j] = __builtin_amdgcn_wmma_f32_16x16x32_f16(
                /*neg_a=*/false, a, /*neg_b=*/false, b,
                /*c_mod=*/(short)0, acc[j], /*reuse_a=*/false, /*reuse_b=*/false);
        }
    }
    int row0 = tm * 16 + ((lane >> 4) << 3);
#pragma unroll
    for (int j = 0; j < NT; j++) {
        int col  = (tn0 + j) * 16 + mn;
        float bs = bias ? bias[col] : 0.f;
#pragma unroll
        for (int r = 0; r < 8; r++)
            C[(size_t)(row0 + r) * N + col] = acc[j][r] + bs;
    }
}

// ---------------------------------------------------------------------------
// residual += hidden; hn = LayerNorm(residual)*w+b -> f16
// one block (256 thr) per token
// ---------------------------------------------------------------------------
__global__ void k_ln(float* __restrict__ residual, const float* __restrict__ hidden,
                     const float* __restrict__ lw, const float* __restrict__ lb,
                     _Float16* __restrict__ hn_h) {
    __shared__ float red[256];
    int t = blockIdx.x, i = threadIdx.x;
    float v = 0.f;
    if (i < D_MODEL) {
        v = residual[t * D_MODEL + i] + hidden[t * D_MODEL + i];
        residual[t * D_MODEL + i] = v;
    }
    red[i] = (i < D_MODEL) ? v : 0.f;
    __syncthreads();
    for (int s = 128; s > 0; s >>= 1) { if (i < s) red[i] += red[i + s]; __syncthreads(); }
    float mean = red[0] / (float)D_MODEL;
    __syncthreads();
    red[i] = (i < D_MODEL) ? (v - mean) * (v - mean) : 0.f;
    __syncthreads();
    for (int s = 128; s > 0; s >>= 1) { if (i < s) red[i] += red[i + s]; __syncthreads(); }
    float var = red[0] / (float)D_MODEL;
    if (i < D_MODEL) {
        float hn = (v - mean) * rsqrtf(var + 1e-5f) * lw[i] + lb[i];
        hn_h[t * D_MODEL + i] = (_Float16)hn;
    }
}

// ---------------------------------------------------------------------------
// causal depthwise conv (k=4) + SiLU, operating in scan space (fwd or flipped)
// u[t,c] = silu(sum_j cw[c,j]*xp[map(t-3+j), c] + cb[c])
// ---------------------------------------------------------------------------
__global__ void k_conv_silu(const float* __restrict__ xz, const float* __restrict__ cw,
                            const float* __restrict__ cb, float* __restrict__ u_f,
                            _Float16* __restrict__ u_h, int fwd) {
    int t = blockIdx.x, c = threadIdx.x;       // 512 x 384
    float acc = cb[c];
#pragma unroll
    for (int j = 0; j < D_CONV; j++) {
        int ts = t - (D_CONV - 1) + j;
        if (ts >= 0) {
            int s = fwd ? ts : (LTOK - 1 - ts);
            acc += cw[c * D_CONV + j] * xz[(size_t)s * (2 * D_INNER) + c];
        }
    }
    float u = acc / (1.f + expf(-acc));        // silu
    u_f[(size_t)t * D_INNER + c] = u;
    u_h[(size_t)t * D_INNER + c] = (_Float16)u;
}

// dt = softplus(dbl[:, :12] @ dpw^T + dpb)
__global__ void k_dt(const float* __restrict__ dbl, const float* __restrict__ dpw,
                     const float* __restrict__ dpb, float* __restrict__ dt) {
    int t = blockIdx.x, c = threadIdx.x;       // 512 x 384
    float acc = dpb[c];
#pragma unroll
    for (int r = 0; r < DT_RANK; r++) acc += dbl[t * XPAD + r] * dpw[c * DT_RANK + r];
    dt[(size_t)t * D_INNER + c] = (acc > 20.f) ? acc : log1pf(expf(acc));
}

// ---------------------------------------------------------------------------
// selective scan: 6144 independent (c,n) chains; 16-lane shfl reduce over n.
// writes gated output at original (un-flipped) position, accumulating fwd+bwd.
// ---------------------------------------------------------------------------
__global__ void k_scan(const float* __restrict__ dt, const float* __restrict__ u_f,
                       const float* __restrict__ dbl, const float* __restrict__ xz,
                       const float* __restrict__ Alog, const float* __restrict__ Dvec,
                       float* __restrict__ ycomb, int fwd) {
    int n  = threadIdx.x & 15;
    int lc = threadIdx.x >> 4;                 // 0..23
    int c  = blockIdx.x * 24 + lc;             // 16 blocks x 24 channels
    float A  = -expf(Alog[c * N_STATE + n]);
    float Dv = Dvec[c];
    float h = 0.f;
    for (int t = 0; t < LTOK; t++) {
        float dtv = dt[(size_t)t * D_INNER + c];
        float uv  = u_f[(size_t)t * D_INNER + c];
        float Bv  = dbl[t * XPAD + DT_RANK + n];
        float Cv  = dbl[t * XPAD + DT_RANK + N_STATE + n];
        h = __expf(dtv * A) * h + dtv * uv * Bv;
        float p = h * Cv;
        p += __shfl_xor(p, 1, 32);
        p += __shfl_xor(p, 2, 32);
        p += __shfl_xor(p, 4, 32);
        p += __shfl_xor(p, 8, 32);
        if (n == 0) {
            int pos = fwd ? t : (LTOK - 1 - t);
            float z = xz[(size_t)pos * (2 * D_INNER) + D_INNER + c];
            float g = z / (1.f + expf(-z));
            ycomb[(size_t)pos * D_INNER + c] += (p + uv * Dv) * g;
        }
    }
}

// final LN on token 511 + 2-way head
__global__ void k_final(const float* __restrict__ residual, const float* __restrict__ hidden,
                        const float* __restrict__ nfw, const float* __restrict__ nfb,
                        const float* __restrict__ hw, const float* __restrict__ hb,
                        float* __restrict__ out) {
    __shared__ float red[256];
    int i = threadIdx.x;
    const int t = LTOK - 1;
    float v = (i < D_MODEL) ? residual[t * D_MODEL + i] + hidden[t * D_MODEL + i] : 0.f;
    red[i] = v; __syncthreads();
    for (int s = 128; s > 0; s >>= 1) { if (i < s) red[i] += red[i + s]; __syncthreads(); }
    float mean = red[0] / (float)D_MODEL; __syncthreads();
    red[i] = (i < D_MODEL) ? (v - mean) * (v - mean) : 0.f; __syncthreads();
    for (int s = 128; s > 0; s >>= 1) { if (i < s) red[i] += red[i + s]; __syncthreads(); }
    float var = red[0] / (float)D_MODEL; __syncthreads();
    float f = (i < D_MODEL) ? (v - mean) * rsqrtf(var + 1e-5f) * nfw[i] + nfb[i] : 0.f;
    for (int o = 0; o < 2; o++) {
        red[i] = (i < D_MODEL) ? f * hw[o * D_MODEL + i] : 0.f; __syncthreads();
        for (int s = 128; s > 0; s >>= 1) { if (i < s) red[i] += red[i + s]; __syncthreads(); }
        if (i == 0) out[o] = red[0] + hb[o];
        __syncthreads();
    }
}

// ---------------------------------------------------------------------------
extern "C" void kernel_launch(void* const* d_in, const int* in_sizes, int n_in,
                              void* d_out, int out_size, void* d_ws, size_t ws_size,
                              hipStream_t stream) {
    const float* x       = (const float*)d_in[0];
    const float* patch_w = (const float*)d_in[1];
    const float* patch_b = (const float*)d_in[2];
    const float* ln_w    = (const float*)d_in[3];
    const float* ln_b    = (const float*)d_in[4];
    const float* in_w    = (const float*)d_in[5];
    const float* cf_w    = (const float*)d_in[6];
    const float* cf_b    = (const float*)d_in[7];
    const float* xpf_w   = (const float*)d_in[8];
    const float* dtf_w   = (const float*)d_in[9];
    const float* dtf_b   = (const float*)d_in[10];
    const float* Alog_f  = (const float*)d_in[11];
    const float* D_f     = (const float*)d_in[12];
    const float* cbk_w   = (const float*)d_in[13];
    const float* cbk_b   = (const float*)d_in[14];
    const float* xpb_w   = (const float*)d_in[15];
    const float* dtb_w   = (const float*)d_in[16];
    const float* dtb_b   = (const float*)d_in[17];
    const float* Alog_b  = (const float*)d_in[18];
    const float* D_b     = (const float*)d_in[19];
    const float* out_w   = (const float*)d_in[20];
    const float* nf_w    = (const float*)d_in[21];
    const float* nf_b    = (const float*)d_in[22];
    const float* head_w  = (const float*)d_in[23];
    const float* head_b  = (const float*)d_in[24];
    float* out = (float*)d_out;

    // --- workspace layout (256B aligned) ---
    char* base = (char*)d_ws;
    size_t off = 0;
    auto alloc = [&](size_t bytes) -> void* {
        void* p = base + off;
        off = (off + bytes + 255) & ~(size_t)255;
        return p;
    };
    _Float16* A0     = (_Float16*)alloc((size_t)LTOK * PATCH_ELEMS * 2);
    _Float16* Wp_h   = (_Float16*)alloc((size_t)D_MODEL * PATCH_ELEMS * 2);
    _Float16* inw_h  = (_Float16*)alloc((size_t)DEPTH * 2 * D_INNER * D_MODEL * 2);
    _Float16* outw_h = (_Float16*)alloc((size_t)DEPTH * D_MODEL * D_INNER * 2);
    _Float16* xpf_h  = (_Float16*)alloc((size_t)DEPTH * XPAD * D_INNER * 2);
    _Float16* xpb_h  = (_Float16*)alloc((size_t)DEPTH * XPAD * D_INNER * 2);
    float*    hidden = (float*)alloc((size_t)LTOK * D_MODEL * 4);
    float*    resid  = (float*)alloc((size_t)LTOK * D_MODEL * 4);
    _Float16* hn_h   = (_Float16*)alloc((size_t)LTOK * D_MODEL * 2);
    float*    xz     = (float*)alloc((size_t)LTOK * 2 * D_INNER * 4);
    float*    u_f    = (float*)alloc((size_t)LTOK * D_INNER * 4);
    _Float16* u_h    = (_Float16*)alloc((size_t)LTOK * D_INNER * 2);
    float*    dbl    = (float*)alloc((size_t)LTOK * XPAD * 4);
    float*    dtb_f  = (float*)alloc((size_t)LTOK * D_INNER * 4);
    float*    ycomb  = (float*)alloc((size_t)LTOK * D_INNER * 4);
    _Float16* y_h    = (_Float16*)alloc((size_t)LTOK * D_INNER * 2);
    (void)ws_size;

    const int CV = 256;
    auto cvt = [&](const float* s, _Float16* d, int n) {
        k_cvt_f32_f16<<<(n + CV - 1) / CV, CV, 0, stream>>>(s, d, n);
    };

    // --- one-time (per call) weight prep ---
    k_im2col<<<LTOK, 256, 0, stream>>>(x, A0);
    cvt(patch_w, Wp_h, D_MODEL * PATCH_ELEMS);
    cvt(in_w, inw_h, DEPTH * 2 * D_INNER * D_MODEL);
    cvt(out_w, outw_h, DEPTH * D_MODEL * D_INNER);
    {
        int n = DEPTH * XPAD * D_INNER;
        k_pad_xproj<<<(n + CV - 1) / CV, CV, 0, stream>>>(xpf_w, xpf_h);
        k_pad_xproj<<<(n + CV - 1) / CV, CV, 0, stream>>>(xpb_w, xpb_h);
    }
    k_zero_f32<<<(LTOK * D_MODEL + CV - 1) / CV, CV, 0, stream>>>(resid, LTOK * D_MODEL);

    // patch embedding GEMM: hidden = A0(512x4096) * Wp^T(192x4096) + patch_b
    k_wmma_gemm<4><<<dim3(D_MODEL / 64, LTOK / 16), 32, 0, stream>>>(
        A0, Wp_h, hidden, patch_b, LTOK, D_MODEL, PATCH_ELEMS);

    for (int l = 0; l < DEPTH; l++) {
        // residual += hidden; hn = LN(residual)
        k_ln<<<LTOK, 256, 0, stream>>>(resid, hidden, ln_w + l * D_MODEL,
                                       ln_b + l * D_MODEL, hn_h);
        // xz = hn @ in_w^T   (512 x 768) = (512x192)*(768x192)^T
        k_wmma_gemm<4><<<dim3(2 * D_INNER / 64, LTOK / 16), 32, 0, stream>>>(
            hn_h, inw_h + (size_t)l * 2 * D_INNER * D_MODEL, xz, (const float*)nullptr,
            LTOK, 2 * D_INNER, D_MODEL);

        k_zero_f32<<<(LTOK * D_INNER + CV - 1) / CV, CV, 0, stream>>>(ycomb, LTOK * D_INNER);

        for (int dir = 0; dir < 2; dir++) {
            int fwd = (dir == 0);
            const float* cw  = fwd ? cf_w  + (size_t)l * D_INNER * D_CONV : cbk_w + (size_t)l * D_INNER * D_CONV;
            const float* cb  = fwd ? cf_b  + (size_t)l * D_INNER          : cbk_b + (size_t)l * D_INNER;
            const _Float16* xph = fwd ? xpf_h + (size_t)l * XPAD * D_INNER : xpb_h + (size_t)l * XPAD * D_INNER;
            const float* dpw = fwd ? dtf_w + (size_t)l * D_INNER * DT_RANK : dtb_w + (size_t)l * D_INNER * DT_RANK;
            const float* dpb = fwd ? dtf_b + (size_t)l * D_INNER          : dtb_b + (size_t)l * D_INNER;
            const float* Al  = fwd ? Alog_f + (size_t)l * D_INNER * N_STATE : Alog_b + (size_t)l * D_INNER * N_STATE;
            const float* Dv  = fwd ? D_f   + (size_t)l * D_INNER          : D_b   + (size_t)l * D_INNER;

            k_conv_silu<<<LTOK, D_INNER, 0, stream>>>(xz, cw, cb, u_f, u_h, fwd);
            // dbl(512x48) = u(512x384) @ xproj^T(48x384)  (3 N-tiles per wave)
            k_wmma_gemm<3><<<dim3(XPAD / 48, LTOK / 16), 32, 0, stream>>>(
                u_h, xph, dbl, (const float*)nullptr, LTOK, XPAD, D_INNER);
            k_dt<<<LTOK, D_INNER, 0, stream>>>(dbl, dpw, dpb, dtb_f);
            k_scan<<<D_INNER / 24, D_INNER, 0, stream>>>(dtb_f, u_f, dbl, xz, Al, Dv, ycomb, fwd);
        }

        cvt(ycomb, y_h, LTOK * D_INNER);
        // hidden = ycomb @ out_w^T  (512x192) = (512x384)*(192x384)^T
        k_wmma_gemm<4><<<dim3(D_MODEL / 64, LTOK / 16), 32, 0, stream>>>(
            y_h, outw_h + (size_t)l * D_MODEL * D_INNER, hidden, (const float*)nullptr,
            LTOK, D_MODEL, D_INNER);
    }

    k_final<<<1, 256, 0, stream>>>(resid, hidden, nf_w, nf_b, head_w, head_b, out);
}
